// SharedNotesCrossAttention_46806553592295
// MI455X (gfx1250) — matrix operand order repaired and checked by reading, not compiled, not run on hardware
//
#include <hip/hip_runtime.h>
#include <hip/hip_bf16.h>

#define B_ 4
#define T_ 2048
#define S_ 512
#define HID_ 2048
#define ND_ 1024
#define NH_ 16
#define HD_ 128

typedef __bf16 v16bf __attribute__((ext_vector_type(16)));
typedef float v8f __attribute__((ext_vector_type(8)));
typedef unsigned int u32x4 __attribute__((ext_vector_type(4)));
typedef int i32x8 __attribute__((ext_vector_type(8)));
typedef int i32x4 __attribute__((ext_vector_type(4)));

__device__ __forceinline__ unsigned short f2bf(float f) {
    unsigned int u = __float_as_uint(f);
    unsigned int r = u + 0x7FFFu + ((u >> 16) & 1u);
    return (unsigned short)(r >> 16);
}

union AP {
    v16bf v;
    uint4 q[2];
    unsigned short u[16];
};

__device__ __forceinline__ v8f wmma_bf16(const AP& a, const AP& b, v8f c) {
    return __builtin_amdgcn_wmma_f32_16x16x32_bf16(false, a.v, false, b.v,
                                                   (short)0, c, false, false);
}

// ---------------------------------------------------------------------------
// f32 -> bf16 conversion pass (bf16 WMMA has 8x the K-depth of f32 WMMA)
// ---------------------------------------------------------------------------
__global__ void f32_to_bf16_kernel(const float* __restrict__ src,
                                   unsigned short* __restrict__ dst, int n) {
    int i = blockIdx.x * blockDim.x + threadIdx.x;
    if (i < n) dst[i] = f2bf(src[i]);
}

// ---------------------------------------------------------------------------
// NT GEMM: C[m,n] = sum_k A[m,k]*B[n,k] + bias[n]
// Block = 8 waves -> 128x128 C tile; wave -> 32x64 sub-tile (2m x 4n tiles).
// B slab (128 n-rows x 64 k) staged to LDS by the Tensor Data Mover,
// double-buffered with s_wait_tensorcnt; TDM pad gives a 144B LDS row pitch
// (36 dwords) so per-lane 32B ds reads land on distinct bank groups.
// Inner loop: flattened 8 steps with explicit double-buffered B registers so
// each step's ds loads overlap the previous step's WMMAs.
// ---------------------------------------------------------------------------
#define KSTEP 64
#define TDM_PITCH 144  // 128B row + 16B TDM pad

__device__ __forceinline__ void tdm_load_b_slab(
    const unsigned short* gsrc, unsigned int lds_addr, int K) {
    unsigned long long ga = (unsigned long long)(uintptr_t)gsrc;
    // D# group0: count=1 | lds_addr | global_addr[56:0] | type=2
    u32x4 g0;
    g0.x = 1u;
    g0.y = lds_addr;
    g0.z = (unsigned int)ga;
    g0.w = (unsigned int)(ga >> 32) | (2u << 30);
    // D# group1: data_size=2B, pad_enable, pad_interval=128B, pad_amount=16B,
    // tensor_dim0=64, tensor_dim1=128, tile_dim0=64, tile_dim1=128,
    // tensor_dim0_stride=K
    i32x8 g1;
    g1[0] = (int)((1u << 16) | (1u << 20) | (4u << 22) | (3u << 25));
    g1[1] = (int)(64u << 16);
    g1[2] = (int)(128u << 16);
    g1[3] = (int)(64u << 16);
    g1[4] = 128;
    g1[5] = K;
    g1[6] = 0;
    g1[7] = 0;
    i32x4 z4 = {0, 0, 0, 0};
    i32x8 z8 = {0, 0, 0, 0, 0, 0, 0, 0};
    __builtin_amdgcn_tensor_load_to_lds(g0, g1, z4, z4, z8, 0);
}

__global__ __launch_bounds__(256) void gemm_nt_bf16_kernel(
    const unsigned short* __restrict__ A, const unsigned short* __restrict__ Bm,
    const float* __restrict__ bias, void* __restrict__ C,
    int M, int N, int K, int out_f32) {
    __shared__ __align__(16) unsigned char bstage[2][128 * TDM_PITCH];

    const int lane = threadIdx.x & 31;
    const int wave = threadIdx.x >> 5;
    const int hi = lane >> 4;
    const int ln = lane & 15;
    const int mw = wave & 3;   // 4 m-groups of 32 rows
    const int nw = wave >> 2;  // 2 n-groups of 64 cols
    const int m0 = blockIdx.x * 128 + mw * 32;
    const int n0loc = nw * 64;
    const int n0 = blockIdx.y * 128 + n0loc;

    v8f acc[2][4] = {};
    const unsigned short* Arow0 = A + (size_t)(m0 + ln) * K;
    const unsigned short* Arow1 = A + (size_t)(m0 + 16 + ln) * K;
    const unsigned short* Bslab = Bm + (size_t)n0 * K;  // unused cols ok
    const unsigned short* Bbase = Bm + (size_t)(blockIdx.y * 128) * K;

    if (wave == 0)
        tdm_load_b_slab(Bbase, (unsigned int)(uintptr_t)&bstage[0][0], K);

    const int nk = K / KSTEP;
    for (int kt = 0; kt < nk; ++kt) {
        const int cur = kt & 1;
        if (wave == 0) __builtin_amdgcn_s_wait_tensorcnt(0);
        __syncthreads();  // current slab ready; prior slab reads all retired
        if (wave == 0 && kt + 1 < nk)
            tdm_load_b_slab(Bbase + (kt + 1) * KSTEP,
                            (unsigned int)(uintptr_t)&bstage[cur ^ 1][0], K);

        const int k0 = kt * KSTEP;
        // A: [chunk][m-tile] 16x32 pieces, per-lane two contiguous 16B runs
        AP a00, a01, a10, a11;
        a00.q[0] = *(const uint4*)(Arow0 + k0 + hi * 8);
        a00.q[1] = *(const uint4*)(Arow0 + k0 + 16 + hi * 8);
        a10.q[0] = *(const uint4*)(Arow0 + k0 + 32 + hi * 8);
        a10.q[1] = *(const uint4*)(Arow0 + k0 + 48 + hi * 8);
        a01.q[0] = *(const uint4*)(Arow1 + k0 + hi * 8);
        a01.q[1] = *(const uint4*)(Arow1 + k0 + 16 + hi * 8);
        a11.q[0] = *(const uint4*)(Arow1 + k0 + 32 + hi * 8);
        a11.q[1] = *(const uint4*)(Arow1 + k0 + 48 + hi * 8);
        __builtin_prefetch(Arow0 + k0 + 2 * KSTEP, 0, 3);
        __builtin_prefetch(Arow1 + k0 + 2 * KSTEP, 0, 3);

        // flattened steps: step = c*4 + j  (c = k-chunk, j = n-tile)
        uint4 bb[2][2];
        const unsigned char* slab = &bstage[cur][0];
        auto ldb = [&](int step, int buf) {
            int c = step >> 2, j = step & 3;
            const unsigned char* brow =
                slab + (n0loc + j * 16 + ln) * TDM_PITCH + c * 64 + hi * 16;
            bb[buf][0] = ((const uint4*)brow)[0];
            bb[buf][1] = ((const uint4*)brow)[2];
        };
        ldb(0, 0);
#pragma unroll
        for (int step = 0; step < 8; ++step) {
            if (step + 1 < 8) ldb(step + 1, (step + 1) & 1);
            AP bm;
            bm.q[0] = bb[step & 1][0];
            bm.q[1] = bb[step & 1][1];
            const int c = step >> 2, j = step & 3;
            acc[0][j] = wmma_bf16(c ? a10 : a00, bm, acc[0][j]);
            acc[1][j] = wmma_bf16(c ? a11 : a01, bm, acc[1][j]);
        }
    }
    (void)Bslab;

    // C layout: VGPR i -> M = hi*8 + i, N = lane&15
#pragma unroll
    for (int mi = 0; mi < 2; ++mi) {
#pragma unroll
        for (int j = 0; j < 4; ++j) {
            int n = n0 + j * 16 + ln;
            float bv = bias[n];
#pragma unroll
            for (int i = 0; i < 8; ++i) {
                int m = m0 + mi * 16 + hi * 8 + i;
                float val = acc[mi][j][i] + bv;
                if (out_f32)
                    ((float*)C)[(size_t)m * N + n] = val;
                else
                    ((unsigned short*)C)[(size_t)m * N + n] = f2bf(val);
            }
        }
    }
}

// ---------------------------------------------------------------------------
// V transpose: (b, s, h*HD+d) bf16 -> Vt[(b*NH+h)*HD + d][s]
// ---------------------------------------------------------------------------
__global__ void transpose_v_kernel(const unsigned short* __restrict__ v,
                                   unsigned short* __restrict__ vt, int n) {
    int i = blockIdx.x * blockDim.x + threadIdx.x;
    if (i >= n) return;
    int c = i % HID_;
    int r = i / HID_;
    int s = r % S_;
    int b = r / S_;
    int h = c / HD_;
    int d = c % HD_;
    vt[(((size_t)b * NH_ + h) * HD_ + d) * S_ + s] = v[i];
}

// ---------------------------------------------------------------------------
// Fused attention for one (b, h, 16-row T tile), one wave per workgroup.
// ---------------------------------------------------------------------------
__global__ __launch_bounds__(32) void attn_kernel(
    const unsigned short* __restrict__ Q, const unsigned short* __restrict__ Km,
    const unsigned short* __restrict__ Vt, const unsigned char* __restrict__ mask,
    const float* __restrict__ gate, unsigned short* __restrict__ ctx) {
    __shared__ float sc[16 * S_];
    __shared__ float red[32];

    const int lane = threadIdx.x & 31;
    const int hi = lane >> 4;
    const int ln = lane & 15;
    const int t0 = blockIdx.x * 16;
    const int h = blockIdx.y;
    const int b = blockIdx.z;

    AP a[4];
    const unsigned short* Qrow = Q + (size_t)(b * T_ + t0 + ln) * HID_ + h * HD_;
#pragma unroll
    for (int kc = 0; kc < 4; ++kc) {
        a[kc].q[0] = *(const uint4*)(Qrow + kc * 32 + hi * 8);
        a[kc].q[1] = *(const uint4*)(Qrow + kc * 32 + 16 + hi * 8);
    }

    const float scale = 0.08838834764831845f;  // 1/sqrt(128)
    const unsigned char* mrow = mask + b * S_;
    const unsigned short* Kbase = Km + (size_t)(b * S_ + ln) * HID_ + h * HD_ + hi * 8;

    // ---- scores: 16 x 512, K tiles double-buffered across s-tiles ----
    auto load_k = [&](int st, uint4(&dst)[8]) {
        const unsigned short* Krow = Kbase + (size_t)(st * 16) * HID_;
#pragma unroll
        for (int kc = 0; kc < 4; ++kc) {
            dst[kc * 2 + 0] = ((const uint4*)(Krow + kc * 32))[0];
            dst[kc * 2 + 1] = ((const uint4*)(Krow + kc * 32))[2];
        }
    };
    auto score_tile = [&](int st, uint4(&kb)[8]) {
        v8f c = {};
#pragma unroll
        for (int kc = 0; kc < 4; ++kc) {
            AP bm;
            bm.q[0] = kb[kc * 2];
            bm.q[1] = kb[kc * 2 + 1];
            c = wmma_bf16(a[kc], bm, c);
        }
        int s = st * 16 + ln;
        float madd = mrow[s] ? 0.0f : -__builtin_inff();
#pragma unroll
        for (int i = 0; i < 8; ++i)
            sc[(hi * 8 + i) * S_ + s] = c[i] * scale + madd;
    };
    {
        uint4 kb0[8], kb1[8];
        load_k(0, kb0);
        for (int it = 0; it < S_ / 32; ++it) {
            const int st0 = 2 * it, st1 = 2 * it + 1;
            load_k(st1, kb1);
            score_tile(st0, kb0);
            if (st1 + 1 < S_ / 16) load_k(st1 + 1, kb0);
            score_tile(st1, kb1);
        }
    }
    __syncthreads();

    // ---- softmax; lane pair (ln, ln+16) splits each row ----
    {
        float* row = sc + ln * S_;
        float m = -__builtin_inff();
        for (int s = hi * 256; s < hi * 256 + 256; ++s) m = fmaxf(m, row[s]);
        red[lane] = m;
        __syncthreads();
        m = fmaxf(red[ln], red[ln + 16]);
        __syncthreads();
        float sum = 0.0f;
        for (int s = hi * 256; s < hi * 256 + 256; ++s) {
            float p = __expf(row[s] - m);
            row[s] = p;
            sum += p;
        }
        red[lane] = sum;
        __syncthreads();
        float g = 1.0f / (1.0f + __expf(-gate[0]));  // sigmoid gate folded in
        float inv = g / (red[ln] + red[ln + 16]);
        for (int s = hi * 256; s < hi * 256 + 256; ++s) row[s] *= inv;
    }
    __syncthreads();

    // ---- PV: (16x512) @ (512x128), V tiles double-buffered within s-chunk --
    v8f acc[8] = {};
    const unsigned short* Vbase =
        Vt + ((size_t)(b * NH_ + h) * HD_ + ln) * S_ + hi * 8;
    for (int si = 0; si < S_ / 32; ++si) {
        AP ap;
        const float* prow = sc + ln * S_ + si * 32;
#pragma unroll
        for (int e = 0; e < 8; ++e) ap.u[e] = f2bf(prow[hi * 8 + e]);
#pragma unroll
        for (int e = 0; e < 8; ++e) ap.u[8 + e] = f2bf(prow[16 + hi * 8 + e]);

        uint4 vb0[2], vb1[2];
        auto load_v = [&](int j, uint4(&dst)[2]) {
            const unsigned short* Vrow = Vbase + (size_t)(j * 16) * S_ + si * 32;
            dst[0] = ((const uint4*)Vrow)[0];
            dst[1] = ((const uint4*)Vrow)[2];
        };
        load_v(0, vb0);
#pragma unroll
        for (int jj = 0; jj < 4; ++jj) {
            const int j0 = 2 * jj, j1 = j0 + 1;
            load_v(j1, vb1);
            AP bm0;
            bm0.q[0] = vb0[0];
            bm0.q[1] = vb0[1];
            acc[j0] = wmma_bf16(ap, bm0, acc[j0]);
            if (j1 < 7) load_v(j1 + 1, vb0);
            AP bm1;
            bm1.q[0] = vb1[0];
            bm1.q[1] = vb1[1];
            acc[j1] = wmma_bf16(ap, bm1, acc[j1]);
        }
    }

#pragma unroll
    for (int j = 0; j < 8; ++j) {
#pragma unroll
        for (int i = 0; i < 8; ++i) {
            int t = t0 + hi * 8 + i;
            int d = j * 16 + ln;
            ctx[(size_t)(b * T_ + t) * HID_ + h * HD_ + d] = f2bf(acc[j][i]);
        }
    }
}

// ---------------------------------------------------------------------------
extern "C" void kernel_launch(void* const* d_in, const int* in_sizes, int n_in,
                              void* d_out, int out_size, void* d_ws, size_t ws_size,
                              hipStream_t stream) {
    const float* hs = (const float*)d_in[0];
    const float* notes = (const float*)d_in[1];
    const unsigned char* mask = (const unsigned char*)d_in[2];
    const float* q_w = (const float*)d_in[3];
    const float* q_b = (const float*)d_in[4];
    const float* k_w = (const float*)d_in[5];
    const float* k_b = (const float*)d_in[6];
    const float* v_w = (const float*)d_in[7];
    const float* v_b = (const float*)d_in[8];
    const float* o_w = (const float*)d_in[9];
    const float* o_b = (const float*)d_in[10];
    const float* gate = (const float*)d_in[11];

    char* ws = (char*)d_ws;
    size_t off = 0;
    auto alloc_bf = [&](size_t elems) {
        unsigned short* p = (unsigned short*)(ws + off);
        off += ((elems * 2 + 255) & ~(size_t)255);
        return p;
    };
    unsigned short* hs_bf    = alloc_bf((size_t)B_ * T_ * HID_);
    unsigned short* notes_bf = alloc_bf((size_t)B_ * S_ * ND_);
    unsigned short* qw_bf    = alloc_bf((size_t)HID_ * HID_);
    unsigned short* kw_bf    = alloc_bf((size_t)HID_ * ND_);
    unsigned short* vw_bf    = alloc_bf((size_t)HID_ * ND_);
    unsigned short* ow_bf    = alloc_bf((size_t)HID_ * HID_);
    unsigned short* q_bf     = alloc_bf((size_t)B_ * T_ * HID_);
    unsigned short* k_bf     = alloc_bf((size_t)B_ * S_ * HID_);
    unsigned short* v_bf     = alloc_bf((size_t)B_ * S_ * HID_);
    unsigned short* vt_bf    = alloc_bf((size_t)B_ * S_ * HID_);
    unsigned short* ctx_bf   = alloc_bf((size_t)B_ * T_ * HID_);
    (void)ws_size; (void)n_in; (void)in_sizes; (void)out_size;

    auto conv = [&](const float* s, unsigned short* d, int n) {
        f32_to_bf16_kernel<<<(n + 255) / 256, 256, 0, stream>>>(s, d, n);
    };
    conv(hs, hs_bf, B_ * T_ * HID_);
    conv(notes, notes_bf, B_ * S_ * ND_);
    conv(q_w, qw_bf, HID_ * HID_);
    conv(k_w, kw_bf, HID_ * ND_);
    conv(v_w, vw_bf, HID_ * ND_);
    conv(o_w, ow_bf, HID_ * HID_);

    // Q = hs @ q_w^T + q_b           (8192 x 2048, K=2048)
    gemm_nt_bf16_kernel<<<dim3((B_ * T_) / 128, HID_ / 128), 256, 0, stream>>>(
        hs_bf, qw_bf, q_b, q_bf, B_ * T_, HID_, HID_, 0);
    // K = notes @ k_w^T + k_b        (2048 x 2048, K=1024)
    gemm_nt_bf16_kernel<<<dim3((B_ * S_) / 128, HID_ / 128), 256, 0, stream>>>(
        notes_bf, kw_bf, k_b, k_bf, B_ * S_, HID_, ND_, 0);
    // V = notes @ v_w^T + v_b
    gemm_nt_bf16_kernel<<<dim3((B_ * S_) / 128, HID_ / 128), 256, 0, stream>>>(
        notes_bf, vw_bf, v_b, v_bf, B_ * S_, HID_, ND_, 0);

    {
        int n = B_ * S_ * HID_;
        transpose_v_kernel<<<(n + 255) / 256, 256, 0, stream>>>(v_bf, vt_bf, n);
    }

    attn_kernel<<<dim3(T_ / 16, NH_, B_), 32, 0, stream>>>(
        q_bf, k_bf, vt_bf, mask, gate, ctx_bf);

    // delta = ctx @ o_w^T + o_b      (f32 output)
    gemm_nt_bf16_kernel<<<dim3((B_ * T_) / 128, HID_ / 128), 256, 0, stream>>>(
        ctx_bf, ow_bf, o_b, d_out, B_ * T_, HID_, HID_, 1);
}
